// MultiHeadAttention_40973988004202
// MI455X (gfx1250) — compile-verified, matrix operand order checked by reading, not run on hardware
//
#include <hip/hip_runtime.h>

typedef __attribute__((ext_vector_type(16))) __bf16 v16bf;
typedef __attribute__((ext_vector_type(8)))  __bf16 v8bf;
typedef __attribute__((ext_vector_type(8)))  float  v8f;
typedef __attribute__((ext_vector_type(4)))  float  v4f;

#define D_MODEL 1024
#define S_LEN   2048
#define HEADS   16
#define DK      64
#define LOG2E   1.4426950408889634f

// xor-shuffle within 16-lane groups via ds_swizzle (group-of-32 mode):
// new_lane = (lane & 0x1f) ^ m   (m < 16 never crosses the 16-lane halves)
#define SWZ_F(x, m) __int_as_float(__builtin_amdgcn_ds_swizzle(__float_as_int(x), (0x1f | ((m) << 10))))

__device__ __forceinline__ v16bf cat16(v8bf lo, v8bf hi) {
    return __builtin_shufflevector(lo, hi, 0,1,2,3,4,5,6,7,8,9,10,11,12,13,14,15);
}

// load 8 consecutive f32 and convert to 8 bf16
__device__ __forceinline__ v8bf cvt8(const float* __restrict__ p) {
    v4f a = *(const v4f*)p;
    v4f b = *(const v4f*)(p + 4);
    v8bf r;
    r[0]=(__bf16)a[0]; r[1]=(__bf16)a[1]; r[2]=(__bf16)a[2]; r[3]=(__bf16)a[3];
    r[4]=(__bf16)b[0]; r[5]=(__bf16)b[1]; r[6]=(__bf16)b[2]; r[7]=(__bf16)b[3];
    return r;
}

__device__ __forceinline__ v8bf ld8bf(const __bf16* p) {
    return *(const v8bf*)p;
}

// 32-bit LDS byte offset of a __shared__ object (addrspace(3) pointers are 32-bit)
__device__ __forceinline__ unsigned lds_off(const void* p) {
    return (unsigned)(unsigned long long)(const __attribute__((address_space(3))) void*)p;
}

// one-lane async copy of 16 bytes global -> LDS (gfx1250 ASYNCcnt path)
__device__ __forceinline__ void async_ld16(unsigned lds, const void* g) {
    unsigned long long ga = (unsigned long long)g;
    asm volatile("global_load_async_to_lds_b128 %0, %1, off"
                 :: "v"(lds), "v"(ga) : "memory");
}

__device__ __forceinline__ void wait_async0() {
    asm volatile("s_wait_asynccnt 0x0" ::: "memory");
}

// ---------------------------------------------------------------------------
// Bulk f32 -> bf16 conversion (one pass; keeps conversions out of GEMM loops)
// ---------------------------------------------------------------------------
__global__ __launch_bounds__(256)
void cvt_kernel(const float* __restrict__ src, unsigned short* __restrict__ dst, int n)
{
    int i = (blockIdx.x * 256 + threadIdx.x) * 8;
    if (i >= n) return;
    *(v8bf*)((__bf16*)dst + i) = cvt8(src + i);
}

// ---------------------------------------------------------------------------
// Projection GEMM (bf16 inputs): C[m,n] = sum_k X[m,k]*W[n,k] + bias[n].
// mode 0: out[b,h,s,d] bf16 (Q scale=1/8, K scale=1)
// mode 2: out[b,h,d,s] bf16 (V transposed for contiguous PV B-fragments)
// Block 128 thr = 4 waves; block tile 128x128; wave tile 64x64.
// ---------------------------------------------------------------------------
__global__ __launch_bounds__(128)
void proj_kernel(const unsigned short* __restrict__ X_, const unsigned short* __restrict__ W_,
                 const float* __restrict__ bias, unsigned short* __restrict__ out_,
                 int mode, float scale)
{
    const __bf16* X = (const __bf16*)X_;
    const __bf16* W = (const __bf16*)W_;
    __bf16* out = (__bf16*)out_;
    const int lane  = threadIdx.x & 31;
    const int wid   = threadIdx.x >> 5;
    const int m0    = blockIdx.x * 128 + (wid >> 1) * 64;
    const int n0    = blockIdx.y * 128 + (wid & 1) * 64;
    const int lrow  = lane & 15;
    const int khalf = (lane >> 4) * 8;   // per-lane K offset within fragment
    const int lhalf = khalf;             // per-lane row offset in C layout

    v8f acc[4][4] = {};
    for (int kk = 0; kk < D_MODEL; kk += 32) {
        v16bf a[4], b[4];
#pragma unroll
        for (int mi = 0; mi < 4; ++mi) {
            const __bf16* p = X + (size_t)(m0 + mi*16 + lrow) * D_MODEL + kk + khalf;
            a[mi] = cat16(ld8bf(p), ld8bf(p + 16));
        }
#pragma unroll
        for (int ni = 0; ni < 4; ++ni) {
            const __bf16* p = W + (size_t)(n0 + ni*16 + lrow) * D_MODEL + kk + khalf;
            b[ni] = cat16(ld8bf(p), ld8bf(p + 16));
        }
#pragma unroll
        for (int mi = 0; mi < 4; ++mi)
#pragma unroll
            for (int ni = 0; ni < 4; ++ni)
                acc[mi][ni] = __builtin_amdgcn_wmma_f32_16x16x32_bf16(
                    false, a[mi], false, b[ni], (short)0, acc[mi][ni], false, false);
    }

#pragma unroll
    for (int ni = 0; ni < 4; ++ni) {
        const int n  = n0 + ni*16 + lrow;
        const float bv = bias[n];
        const int h = n >> 6, d = n & 63;
#pragma unroll
        for (int mi = 0; mi < 4; ++mi) {
#pragma unroll
            for (int r = 0; r < 8; ++r) {
                const int m  = m0 + mi*16 + r + lhalf;
                const int bb = m >> 11;       // batch (S=2048)
                const int s  = m & 2047;
                const float v = (acc[mi][ni][r] + bv) * scale;
                size_t addr;
                if (mode == 2)
                    addr = (((size_t)(bb*HEADS + h))*DK + d)*S_LEN + s;   // [b,h,d,s]
                else
                    addr = (((size_t)(bb*HEADS + h))*S_LEN + s)*DK + d;   // [b,h,s,d]
                out[addr] = (__bf16)v;
            }
        }
    }
}

// ---------------------------------------------------------------------------
// Flash attention: block = 4 waves x 32 q-rows = 128 q rows of one (b,h).
// K/V tiles (64 keys) double-buffered in LDS via async global->LDS copies,
// shared by all 4 waves. Q:[bh,S,64] (pre-scaled), K:[bh,S,64], Vt:[bh,64,S].
// ---------------------------------------------------------------------------
__global__ __launch_bounds__(128)
void flash_kernel(const unsigned short* __restrict__ Q_,
                  const unsigned short* __restrict__ K_,
                  const unsigned short* __restrict__ Vt_,
                  unsigned short* __restrict__ attn_)
{
    const __bf16* Q  = (const __bf16*)Q_;
    const __bf16* K  = (const __bf16*)K_;
    const __bf16* Vt = (const __bf16*)Vt_;
    __bf16* attn = (__bf16*)attn_;

    __shared__ __align__(16) __bf16 kbuf[2][64 * 64];   // [kv row][d]      8KB x2
    __shared__ __align__(16) __bf16 vbuf[2][64 * 64];   // [d][kv col]      8KB x2
    __shared__ __align__(16) __bf16 pbuf[4][32 * 64];   // per-wave P tile  4KB x4

    const int tid   = threadIdx.x;
    const int lane  = tid & 31;
    const int wid   = tid >> 5;
    const int lrow  = lane & 15;
    const int khalf = (lane >> 4) * 8;
    const int lhalf = khalf;
    const int bh = blockIdx.y;                      // b*16 + h
    const int q0 = blockIdx.x * 128 + wid * 32;     // 32 q rows per wave
    const __bf16* Qh = Q  + (size_t)bh * S_LEN * DK;
    const __bf16* Kh = K  + (size_t)bh * S_LEN * DK;
    const __bf16* Vh = Vt + (size_t)bh * DK * S_LEN;

    // Q A-fragments: 2 m-subtiles x 2 d-chunks, kept in registers
    v16bf qa[2][2];
#pragma unroll
    for (int mi = 0; mi < 2; ++mi)
#pragma unroll
        for (int kc = 0; kc < 2; ++kc) {
            const __bf16* p = Qh + (size_t)(q0 + mi*16 + lrow) * DK + kc*32 + khalf;
            qa[mi][kc] = cat16(ld8bf(p), ld8bf(p + 16));
        }

    v8f o[2][4] = {};
    float mrun[2][8], lrun[2][8];
#pragma unroll
    for (int mi = 0; mi < 2; ++mi)
#pragma unroll
        for (int r = 0; r < 8; ++r) { mrun[mi][r] = -1e30f; lrun[mi][r] = 0.0f; }

    // cooperative async stage of one 64-key K tile + V tile into buffer `buf`
    auto stage = [&](int buf, int j) {
#pragma unroll
        for (int i = 0; i < 4; ++i) {
            const int c   = tid + i * 128;      // 512 chunks of 16B per tile
            const int row = c >> 3;
            const int col = (c & 7) * 8;
            async_ld16(lds_off(&kbuf[buf][row * 64 + col]),
                       Kh + (size_t)(j + row) * DK + col);
            async_ld16(lds_off(&vbuf[buf][row * 64 + col]),
                       Vh + (size_t)row * S_LEN + j + col);
        }
    };

    stage(0, 0);

    const int NTILES = S_LEN / 64;
    for (int jt = 0; jt < NTILES; ++jt) {
        const int cur = jt & 1;
        wait_async0();
        __syncthreads();                       // staged K/V visible to all waves
        if (jt + 1 < NTILES) stage(cur ^ 1, (jt + 1) * 64);

        // ---- scores: S = Q * K^T (32 x 64 per wave) ----
        v8f s[2][4] = {};
#pragma unroll
        for (int kc = 0; kc < 2; ++kc) {
#pragma unroll
            for (int ni = 0; ni < 4; ++ni) {
                const __bf16* p = &kbuf[cur][(ni*16 + lrow) * 64 + kc*32 + khalf];
                v16bf kb = cat16(ld8bf(p), ld8bf(p + 16));
#pragma unroll
                for (int mi = 0; mi < 2; ++mi)
                    s[mi][ni] = __builtin_amdgcn_wmma_f32_16x16x32_bf16(
                        false, qa[mi][kc], false, kb, (short)0, s[mi][ni], false, false);
            }
        }

        // ---- online softmax ----
#pragma unroll
        for (int mi = 0; mi < 2; ++mi) {
            float mnew[8], corr[8];
#pragma unroll
            for (int r = 0; r < 8; ++r) {
                float t = fmaxf(fmaxf(s[mi][0][r], s[mi][1][r]),
                                fmaxf(s[mi][2][r], s[mi][3][r]));
                t = fmaxf(t, SWZ_F(t, 1));
                t = fmaxf(t, SWZ_F(t, 2));
                t = fmaxf(t, SWZ_F(t, 4));
                t = fmaxf(t, SWZ_F(t, 8));
                mnew[r] = fmaxf(mrun[mi][r], t);
                corr[r] = exp2f((mrun[mi][r] - mnew[r]) * LOG2E);
                mrun[mi][r] = mnew[r];
            }
            float rs[8];
#pragma unroll
            for (int r = 0; r < 8; ++r) rs[r] = 0.0f;
#pragma unroll
            for (int ni = 0; ni < 4; ++ni) {
#pragma unroll
                for (int r = 0; r < 8; ++r) {
                    float pv = exp2f((s[mi][ni][r] - mnew[r]) * LOG2E);
                    rs[r] += pv;
                    pbuf[wid][(mi*16 + r + lhalf) * 64 + ni*16 + lrow] = (__bf16)pv;
                }
            }
#pragma unroll
            for (int r = 0; r < 8; ++r) {
                float t = rs[r];
                t += SWZ_F(t, 1);
                t += SWZ_F(t, 2);
                t += SWZ_F(t, 4);
                t += SWZ_F(t, 8);
                lrun[mi][r] = lrun[mi][r] * corr[r] + t;
#pragma unroll
                for (int ni = 0; ni < 4; ++ni) o[mi][ni][r] *= corr[r];
            }
        }

        asm volatile("s_wait_dscnt 0x0" ::: "memory");

        // ---- O += P * V ----
#pragma unroll
        for (int kc = 0; kc < 2; ++kc) {
            v16bf pa[2];
#pragma unroll
            for (int mi = 0; mi < 2; ++mi) {
                const __bf16* pp = &pbuf[wid][(mi*16 + lrow) * 64 + kc*32 + khalf];
                pa[mi] = cat16(ld8bf(pp), ld8bf(pp + 16));
            }
#pragma unroll
            for (int ni = 0; ni < 4; ++ni) {
                const __bf16* vp = &vbuf[cur][(ni*16 + lrow) * 64 + kc*32 + khalf];
                v16bf vb = cat16(ld8bf(vp), ld8bf(vp + 16));
#pragma unroll
                for (int mi = 0; mi < 2; ++mi)
                    o[mi][ni] = __builtin_amdgcn_wmma_f32_16x16x32_bf16(
                        false, pa[mi], false, vb, (short)0, o[mi][ni], false, false);
            }
        }

        __syncthreads();                // all waves done with buffer `cur`
    }

    // ---- finalize: O / l, store bf16 into [B,S,1024] ----
    const int h = bh & 15, b = bh >> 4;
#pragma unroll
    for (int mi = 0; mi < 2; ++mi)
#pragma unroll
        for (int ni = 0; ni < 4; ++ni)
#pragma unroll
            for (int r = 0; r < 8; ++r) {
                const int qrow = q0 + mi*16 + r + lhalf;
                const int d    = ni*16 + lrow;
                float v = o[mi][ni][r] / lrun[mi][r];
                attn[((size_t)(b * S_LEN + qrow)) * D_MODEL + h*DK + d] = (__bf16)v;
            }
}

// ---------------------------------------------------------------------------
// Output projection: out[m,n] = sum_k A_bf16[m,k]*Wo_bf16[n,k] + bo[n] (f32)
// ---------------------------------------------------------------------------
__global__ __launch_bounds__(128)
void oproj_kernel(const unsigned short* __restrict__ A_, const unsigned short* __restrict__ W_,
                  const float* __restrict__ bias, float* __restrict__ out)
{
    const __bf16* A = (const __bf16*)A_;
    const __bf16* W = (const __bf16*)W_;
    const int lane  = threadIdx.x & 31;
    const int wid   = threadIdx.x >> 5;
    const int m0    = blockIdx.x * 128 + (wid >> 1) * 64;
    const int n0    = blockIdx.y * 128 + (wid & 1) * 64;
    const int lrow  = lane & 15;
    const int khalf = (lane >> 4) * 8;
    const int lhalf = khalf;

    v8f acc[4][4] = {};
    for (int kk = 0; kk < D_MODEL; kk += 32) {
        v16bf a[4], b[4];
#pragma unroll
        for (int mi = 0; mi < 4; ++mi) {
            const __bf16* p = A + (size_t)(m0 + mi*16 + lrow) * D_MODEL + kk + khalf;
            a[mi] = cat16(ld8bf(p), ld8bf(p + 16));
        }
#pragma unroll
        for (int ni = 0; ni < 4; ++ni) {
            const __bf16* p = W + (size_t)(n0 + ni*16 + lrow) * D_MODEL + kk + khalf;
            b[ni] = cat16(ld8bf(p), ld8bf(p + 16));
        }
#pragma unroll
        for (int mi = 0; mi < 4; ++mi)
#pragma unroll
            for (int ni = 0; ni < 4; ++ni)
                acc[mi][ni] = __builtin_amdgcn_wmma_f32_16x16x32_bf16(
                    false, a[mi], false, b[ni], (short)0, acc[mi][ni], false, false);
    }

#pragma unroll
    for (int ni = 0; ni < 4; ++ni) {
        const int n = n0 + ni*16 + lrow;
        const float bv = bias[n];
#pragma unroll
        for (int mi = 0; mi < 4; ++mi) {
#pragma unroll
            for (int r = 0; r < 8; ++r) {
                const int m = m0 + mi*16 + r + lhalf;
                out[(size_t)m * D_MODEL + n] = acc[mi][ni][r] + bv;
            }
        }
    }
}

// ---------------------------------------------------------------------------
extern "C" void kernel_launch(void* const* d_in, const int* in_sizes, int n_in,
                              void* d_out, int out_size, void* d_ws, size_t ws_size,
                              hipStream_t stream)
{
    const float* query = (const float*)d_in[0];
    const float* key   = (const float*)d_in[1];
    const float* value = (const float*)d_in[2];
    const float* Wq    = (const float*)d_in[3];
    const float* bq    = (const float*)d_in[4];
    const float* Wk    = (const float*)d_in[5];
    const float* bk    = (const float*)d_in[6];
    const float* Wv    = (const float*)d_in[7];
    const float* bv    = (const float*)d_in[8];
    const float* Wo    = (const float*)d_in[9];
    const float* bo    = (const float*)d_in[10];
    float* out = (float*)d_out;

    char* ws = (char*)d_ws;
    const size_t MB = (size_t)1 << 20;
    unsigned short* Qb  = (unsigned short*)(ws);            // 8 MB [b,h,s,d] bf16
    unsigned short* Kb  = (unsigned short*)(ws + 8*MB);     // 8 MB [b,h,s,d] bf16
    unsigned short* Vt  = (unsigned short*)(ws + 16*MB);    // 8 MB [b,h,d,s] bf16
    unsigned short* Ab  = (unsigned short*)(ws + 24*MB);    // 8 MB [b,s,1024] bf16
    unsigned short* Xq  = (unsigned short*)(ws + 32*MB);    // 8 MB query bf16
    unsigned short* Xk  = (unsigned short*)(ws + 40*MB);    // 8 MB key bf16
    unsigned short* Xv  = (unsigned short*)(ws + 48*MB);    // 8 MB value bf16
    unsigned short* Wqb = (unsigned short*)(ws + 56*MB);    // 2 MB each
    unsigned short* Wkb = (unsigned short*)(ws + 58*MB);
    unsigned short* Wvb = (unsigned short*)(ws + 60*MB);
    unsigned short* Wob = (unsigned short*)(ws + 62*MB);

    const int NX = 2 * S_LEN * D_MODEL;   // 4194304
    const int NW = D_MODEL * D_MODEL;     // 1048576

    cvt_kernel<<<NX/8/256, 256, 0, stream>>>(query, Xq, NX);
    cvt_kernel<<<NX/8/256, 256, 0, stream>>>(key,   Xk, NX);
    cvt_kernel<<<NX/8/256, 256, 0, stream>>>(value, Xv, NX);
    cvt_kernel<<<NW/8/256, 256, 0, stream>>>(Wq, Wqb, NW);
    cvt_kernel<<<NW/8/256, 256, 0, stream>>>(Wk, Wkb, NW);
    cvt_kernel<<<NW/8/256, 256, 0, stream>>>(Wv, Wvb, NW);
    cvt_kernel<<<NW/8/256, 256, 0, stream>>>(Wo, Wob, NW);

    dim3 gProj(32, 8);  // M=4096/128, N=1024/128
    dim3 bProj(128);
    proj_kernel<<<gProj, bProj, 0, stream>>>(Xq, Wqb, bq, Qb, 0, 0.125f); // 1/sqrt(64) folded into Q
    proj_kernel<<<gProj, bProj, 0, stream>>>(Xk, Wkb, bk, Kb, 0, 1.0f);
    proj_kernel<<<gProj, bProj, 0, stream>>>(Xv, Wvb, bv, Vt, 2, 1.0f);   // transposed V

    flash_kernel<<<dim3(S_LEN / 128, 32), dim3(128), 0, stream>>>(Qb, Kb, Vt, Ab);

    oproj_kernel<<<gProj, bProj, 0, stream>>>(Ab, Wob, bo, out);
}
